// LlamaAttention_32547262169190
// MI455X (gfx1250) — matrix operand order repaired and checked by reading, not compiled, not run on hardware
//
#include <hip/hip_runtime.h>

// ---------------- types / helpers ----------------
typedef unsigned short bf16_t;                                  // bf16 storage
typedef __attribute__((ext_vector_type(16))) __bf16 v16bf;      // WMMA A/B frag
typedef __attribute__((ext_vector_type(8)))  float  v8f;        // WMMA C/D frag
typedef unsigned int v4u __attribute__((ext_vector_type(4)));
typedef int v8i __attribute__((ext_vector_type(8)));
typedef int v4i __attribute__((ext_vector_type(4)));

#define DEV static __device__ __forceinline__

DEV float bf2f(bf16_t b) {
    unsigned u = ((unsigned)b) << 16;
    float f; __builtin_memcpy(&f, &u, 4);
    return f;
}
DEV bf16_t f2bf(float f) {
    unsigned u; __builtin_memcpy(&u, &f, 4);
    u += 0x7FFFu + ((u >> 16) & 1u);          // round-to-nearest-even
    return (bf16_t)(u >> 16);
}
DEV v16bf load_frag(const bf16_t* p0, const bf16_t* p1) {
    union { v16bf v; uint4 u[2]; } f;
    f.u[0] = *reinterpret_cast<const uint4*>(p0);
    f.u[1] = *reinterpret_cast<const uint4*>(p1);
    return f.v;
}

// ---- Tensor Data Mover: 2D tile -> LDS (D# per CDNA5 ISA ch.8) ----
// data_size = 8B units. tile_d0/stride0/tdim0 in 8-byte units.
// Optional LDS row padding: pad_interval code c -> pad after 2^(c+1) DWORDs,
// pad_amount code p -> insert (p+1) DWORDs.
DEV void tdm_load_2d(unsigned lds_off, unsigned long long gaddr,
                     unsigned tile_d0, unsigned tile_d1,
                     unsigned stride0, unsigned tdim0, unsigned tdim1,
                     unsigned pad_interval, unsigned pad_amount, unsigned pad_en)
{
    v4u g0;
    g0[0] = 1u;                                        // count=1 (valid D#)
    g0[1] = lds_off;                                   // LDS byte address
    g0[2] = (unsigned)gaddr;                           // global addr [31:0]
    g0[3] = (unsigned)(gaddr >> 32) | (2u << 30);      // addr[56:32] | type=2
    v8i g1;
    g1[0] = (int)((3u << 16) | (pad_en << 20) | (pad_interval << 22) | (pad_amount << 25));
    g1[1] = (int)((tdim0 & 0xFFFFu) << 16);            // tensor_dim0 lo16
    g1[2] = (int)((tdim0 >> 16) | ((tdim1 & 0xFFFFu) << 16));
    g1[3] = (int)((tdim1 >> 16) | (tile_d0 << 16));    // tile_dim0
    g1[4] = (int)(tile_d1 & 0xFFFFu);                  // tile_dim1 (tile_dim2=0)
    g1[5] = (int)stride0;                              // tensor_dim0_stride lo32
    g1[6] = 0;                                         // stride0 hi16 | stride1 lo16
    g1[7] = 0;
    v4i z4 = {0, 0, 0, 0};
    v8i z8 = {0, 0, 0, 0, 0, 0, 0, 0};
    __builtin_amdgcn_tensor_load_to_lds(g0, g1, z4, z4, z8, 0);
}
DEV unsigned lds_off_of(const void* p) {               // LDS flat addr low 32 = offset
    return (unsigned)(unsigned long long)p;
}

// model constants
#define SN    2048      // sequence length
#define SD    4096      // d_model
#define NH    32        // query heads
#define NKV   8         // kv heads
#define HD    128       // head dim
#define QD    (NH*HD)   // 4096
#define KD    (NKV*HD)  // 1024

// ---------------- 1) RMSNorm -> bf16 ----------------
__global__ __launch_bounds__(256) void rmsnorm_bf16_kernel(
    const float* __restrict__ x, const float* __restrict__ w,
    bf16_t* __restrict__ h)
{
    const int row = blockIdx.x;
    const float* xr = x + (size_t)row * SD;
    float ss = 0.f;
    for (int i = threadIdx.x; i < SD; i += 256) { float v = xr[i]; ss += v * v; }
    __shared__ float red[256];
    red[threadIdx.x] = ss;
    __syncthreads();
    for (int s = 128; s > 0; s >>= 1) {
        if (threadIdx.x < s) red[threadIdx.x] += red[threadIdx.x + s];
        __syncthreads();
    }
    const float scale = rsqrtf(red[0] / (float)SD + 1e-5f);
    bf16_t* hr = h + (size_t)row * SD;
    for (int i = threadIdx.x; i < SD; i += 256)
        hr[i] = f2bf(xr[i] * scale * w[i]);
}

// ---------------- 2) fp32 -> bf16 transposed weight convert ----------------
__global__ __launch_bounds__(256) void transpose_f32_bf16_kernel(
    const float* __restrict__ src, bf16_t* __restrict__ dst, int K, int Nout)
{
    __shared__ float tile[32][33];
    const int tx = threadIdx.x & 31;
    const int ty = threadIdx.x >> 5;
    const int k0 = blockIdx.y * 32;
    const int n0 = blockIdx.x * 32;
#pragma unroll
    for (int i = 0; i < 4; i++)
        tile[ty + i * 8][tx] = src[(size_t)(k0 + ty + i * 8) * Nout + n0 + tx];
    __syncthreads();
#pragma unroll
    for (int i = 0; i < 4; i++)
        dst[(size_t)(n0 + ty + i * 8) * K + k0 + tx] = f2bf(tile[tx][ty + i * 8]);
}

// ---------------- 3) WMMA bf16 GEMM with TDM double-buffered LDS staging -----
// C[M x Nn] = A[M x K] * Bt[Nn x K]^T. Block 256 thr = 8 waves.
// Block tile 64(M) x 128(N); wave tile 16 x 64; K step 32.
// LDS rows padded 64B->80B (pad codes 3,3) so ds_load_b128 frags are conflict-free.
// MODE 0: fp32 out + residual (output projection)
// MODE 1: bf16 out with fused RoPE (q/k projections; RoPE pair via shfl_xor 1)
// MODE 2: bf16 out plain (v projection)
#define ASTR 40   // bf16 elems per padded LDS row (32 data + 8 pad)
template <int MODE>
__global__ __launch_bounds__(256) void gemm_bf16_wmma_kernel(
    const bf16_t* __restrict__ A, const bf16_t* __restrict__ Bt,
    const float* __restrict__ residual,
    const float* __restrict__ rc, const float* __restrict__ rs,
    float* __restrict__ Cf, bf16_t* __restrict__ Cb,
    int M, int Nn, int K)
{
    __shared__ bf16_t As[2][64 * ASTR];
    __shared__ bf16_t Bs[2][128 * ASTR];

    const int lane  = threadIdx.x & 31;
    const int wave  = threadIdx.x >> 5;
    const int waveM = wave & 3;
    const int waveN = wave >> 2;
    const int m0b = blockIdx.y * 64;
    const int n0b = blockIdx.x * 128;
    const int half = lane >> 4;
    const int lr   = lane & 15;

    auto issue = [&](int buf, int kk) {
        tdm_load_2d(lds_off_of(&As[buf][0]),
                    (unsigned long long)(A + (size_t)m0b * K + kk),
                    8, 64, (unsigned)(K / 4), (unsigned)(K / 4), (unsigned)M,
                    3, 3, 1);
        tdm_load_2d(lds_off_of(&Bs[buf][0]),
                    (unsigned long long)(Bt + (size_t)n0b * K + kk),
                    8, 128, (unsigned)(K / 4), (unsigned)(K / 4), (unsigned)Nn,
                    3, 3, 1);
    };

    v8f acc[4];
#pragma unroll
    for (int j = 0; j < 4; j++) acc[j] = (v8f){0.f,0.f,0.f,0.f,0.f,0.f,0.f,0.f};

    if (wave == 0) {
        issue(0, 0);
        __builtin_amdgcn_s_wait_tensorcnt(0);
    }
    __syncthreads();

    const int nk = K / 32;
    for (int t = 0; t < nk; t++) {
        const int cur = t & 1;
        if (wave == 0 && t + 1 < nk) issue(cur ^ 1, (t + 1) * 32);

        // batch all DS fragment loads, then issue WMMAs back-to-back
        const bf16_t* Ab = &As[cur][0] + (waveM * 16 + lr) * ASTR;
        v16bf af = load_frag(Ab + 16 * half, Ab + 16 * half + 8);
        v16bf bfr[4];
#pragma unroll
        for (int j = 0; j < 4; j++) {
            const bf16_t* Bb = &Bs[cur][0] + (waveN * 64 + j * 16 + lr) * ASTR;
            bfr[j] = load_frag(Bb + 16 * half, Bb + 16 * half + 8);
        }
#pragma unroll
        for (int j = 0; j < 4; j++)
            acc[j] = __builtin_amdgcn_wmma_f32_16x16x32_bf16(
                false, af, false, bfr[j], (short)0, acc[j], false, false);

        if (wave == 0 && t + 1 < nk) __builtin_amdgcn_s_wait_tensorcnt(0);
        __syncthreads();
    }

    const int m0 = m0b + waveM * 16;
    const int n0 = n0b + waveN * 64;
#pragma unroll
    for (int j = 0; j < 4; j++) {
        const int n = n0 + j * 16 + lr;
#pragma unroll
        for (int r = 0; r < 8; r++) {
            const int m = m0 + r + 8 * half;
            float v = acc[j][r];
            if (MODE == 0) {
                if (residual) v += residual[(size_t)m * Nn + n];
                Cf[(size_t)m * Nn + n] = v;
            } else if (MODE == 1) {
                // fused RoPE: pair element lives in adjacent lane (n ^ 1)
                const float partner = __shfl_xor(v, 1, 32);
                const int d = n & (HD - 1);
                const float c = rc[(size_t)m * HD + d];
                const float s = rs[(size_t)m * HD + d];
                const float o = v * c + ((n & 1) ? partner : -partner) * s;
                Cb[(size_t)m * Nn + n] = f2bf(o);
            } else {
                Cb[(size_t)m * Nn + n] = f2bf(v);
            }
        }
    }
}

// ---------------- 4) WMMA flash attention -----------------------------------
// Block = (head, 128 query rows) = 8 waves x 16 rows. Per 32-key step:
// TDM loads K-tile (padded) and V-tile into LDS; V transposed in LDS;
// QK^T: 8 WMMAs; online softmax (shfl within 16-lane half); P via LDS
// (s_wait_dscnt orders cross-lane DS write->read); PV: 8 WMMAs.
#define KSTR 136            // K-tile LDS row: 128 data + 8 pad bf16 (272B)
#define VSTR 40             // Vt / P LDS row: 32 data + 8 pad bf16 (80B)
__global__ __launch_bounds__(256) void attention_wmma_kernel(
    const bf16_t* __restrict__ qb, const bf16_t* __restrict__ kb,
    const bf16_t* __restrict__ vb, bf16_t* __restrict__ ob)
{
    __shared__ bf16_t Kt[32 * KSTR];
    __shared__ bf16_t Vraw[32 * 128];
    __shared__ bf16_t Vt[128 * VSTR];
    __shared__ bf16_t Pl[8][16 * VSTR];

    const int head = blockIdx.x;
    const int qt   = blockIdx.y;
    const int kvh  = head >> 2;           // GQA 4:1
    const int lane = threadIdx.x & 31;
    const int wave = threadIdx.x >> 5;
    const int half = lane >> 4;
    const int lr   = lane & 15;
    const int q0w  = qt * 128 + wave * 16;

    // Q fragments (16 rows x 128 d = 4 K-chunks)
    v16bf qf[4];
    {
        const bf16_t* qrow = qb + (size_t)(q0w + lr) * QD + (size_t)head * HD;
#pragma unroll
        for (int c = 0; c < 4; c++)
            qf[c] = load_frag(qrow + c * 32 + 16 * half, qrow + c * 32 + 16 * half + 8);
    }

    v8f oacc[8];
#pragma unroll
    for (int j = 0; j < 8; j++) oacc[j] = (v8f){0.f,0.f,0.f,0.f,0.f,0.f,0.f,0.f};
    float mrow[8], lrow[8];
#pragma unroll
    for (int r = 0; r < 8; r++) { mrow[r] = -1e30f; lrow[r] = 0.f; }

    const float scale = 0.08838834764831845f;   // 1/sqrt(128)
    const int jmax = qt * 128 + 127;

    for (int j0 = 0; j0 <= jmax; j0 += 32) {
        if (wave == 0) {
            const unsigned long long kg =
                (unsigned long long)(kb + (size_t)j0 * KD + (size_t)kvh * HD);
            const unsigned long long vg =
                (unsigned long long)(vb + (size_t)j0 * KD + (size_t)kvh * HD);
            tdm_load_2d(lds_off_of(&Kt[0]),   kg, 32, 32, KD / 4, KD / 4, SN, 5, 3, 1);
            tdm_load_2d(lds_off_of(&Vraw[0]), vg, 32, 32, KD / 4, KD / 4, SN, 0, 0, 0);
            __builtin_amdgcn_s_wait_tensorcnt(0);
        }
        __syncthreads();

        // transpose V 32x128 -> Vt[128][VSTR]
        for (int idx = threadIdx.x; idx < 32 * 128; idx += 256) {
            const int key = idx >> 7, d = idx & 127;
            Vt[d * VSTR + key] = Vraw[idx];
        }
        __syncthreads();

        if (j0 <= q0w + 15) {               // wave-uniform causal skip
            // ---- S = Q K^T (two 16-key tiles) ----
            v8f s0 = (v8f){0.f,0.f,0.f,0.f,0.f,0.f,0.f,0.f};
            v8f s1 = s0;
#pragma unroll
            for (int c = 0; c < 4; c++) {
                const bf16_t* k0p = &Kt[0] + lr * KSTR + c * 32 + 16 * half;
                const bf16_t* k1p = &Kt[0] + (16 + lr) * KSTR + c * 32 + 16 * half;
                v16bf b0 = load_frag(k0p, k0p + 8);
                v16bf b1 = load_frag(k1p, k1p + 8);
                s0 = __builtin_amdgcn_wmma_f32_16x16x32_bf16(false, qf[c], false, b0, (short)0, s0, false, false);
                s1 = __builtin_amdgcn_wmma_f32_16x16x32_bf16(false, qf[c], false, b1, (short)0, s1, false, false);
            }
            // ---- online softmax + P -> LDS ----
            bf16_t* Pw = &Pl[wave][0];
#pragma unroll
            for (int r = 0; r < 8; r++) {
                const int qrow_ = q0w + 8 * half + r;
                float a0 = (j0 + lr      <= qrow_) ? s0[r] * scale : -1e30f;
                float a1 = (j0 + 16 + lr <= qrow_) ? s1[r] * scale : -1e30f;
                float t = fmaxf(a0, a1);
#pragma unroll
                for (int off = 8; off > 0; off >>= 1) t = fmaxf(t, __shfl_xor(t, off, 32));
                const float mn   = fmaxf(mrow[r], t);
                const float corr = __expf(mrow[r] - mn);
                const float p0   = __expf(a0 - mn);
                const float p1   = __expf(a1 - mn);
                float ps = p0 + p1;
#pragma unroll
                for (int off = 8; off > 0; off >>= 1) ps += __shfl_xor(ps, off, 32);
                lrow[r] = lrow[r] * corr + ps;
                mrow[r] = mn;
                const int prow = 8 * half + r;
                Pw[prow * VSTR + lr]      = f2bf(p0);
                Pw[prow * VSTR + 16 + lr] = f2bf(p1);
#pragma unroll
                for (int j = 0; j < 8; j++) oacc[j][r] *= corr;
            }
            asm volatile("s_wait_dscnt 0" ::: "memory");   // cross-lane DS WAR/RAW
            const bf16_t* pr = Pw + lr * VSTR;
            v16bf pf = load_frag(pr + 16 * half, pr + 16 * half + 8);
            // ---- O += P V ----
#pragma unroll
            for (int j = 0; j < 8; j++) {
                const bf16_t* vr = &Vt[0] + (j * 16 + lr) * VSTR + 16 * half;
                v16bf vf = load_frag(vr, vr + 8);
                oacc[j] = __builtin_amdgcn_wmma_f32_16x16x32_bf16(false, pf, false, vf, (short)0, oacc[j], false, false);
            }
        }
        __syncthreads();
    }

    // epilogue: normalize, store
#pragma unroll
    for (int r = 0; r < 8; r++) {
        const float inv = 1.0f / lrow[r];
        const int row = q0w + 8 * half + r;
        bf16_t* orow = ob + (size_t)row * QD + (size_t)head * HD;
#pragma unroll
        for (int j = 0; j < 8; j++)
            orow[j * 16 + lr] = f2bf(oacc[j][r] * inv);
    }
}

// ---------------- host launcher ----------------
extern "C" void kernel_launch(void* const* d_in, const int* in_sizes, int n_in,
                              void* d_out, int out_size, void* d_ws, size_t ws_size,
                              hipStream_t stream) {
    const float* x  = (const float*)d_in[0];
    const float* rc = (const float*)d_in[1];
    const float* rs = (const float*)d_in[2];
    const float* wn = (const float*)d_in[3];
    const float* wq = (const float*)d_in[4];
    const float* wk = (const float*)d_in[5];
    const float* wv = (const float*)d_in[6];
    const float* wo = (const float*)d_in[7];
    float* out = (float*)d_out;

    char* ws = (char*)d_ws;
    size_t off = 0;
    auto alloc = [&](size_t bytes) -> void* {
        void* p = ws + off;
        off += (bytes + 255) & ~(size_t)255;
        return p;
    };

    bf16_t* hb    = (bf16_t*)alloc((size_t)SN * SD * 2);   // normed activations
    bf16_t* wq_t  = (bf16_t*)alloc((size_t)SD * QD * 2);   // transposed bf16 weights
    bf16_t* wk_t  = (bf16_t*)alloc((size_t)SD * KD * 2);
    bf16_t* wv_t  = (bf16_t*)alloc((size_t)SD * KD * 2);
    bf16_t* wo_t  = (bf16_t*)alloc((size_t)SD * SD * 2);
    bf16_t* q_b   = (bf16_t*)alloc((size_t)SN * QD * 2);   // roped q (bf16)
    bf16_t* k_b   = (bf16_t*)alloc((size_t)SN * KD * 2);   // roped k (bf16)
    bf16_t* v_b   = (bf16_t*)alloc((size_t)SN * KD * 2);   // v (bf16)
    bf16_t* a_b   = (bf16_t*)alloc((size_t)SN * QD * 2);   // attention output

    // 1) RMSNorm
    rmsnorm_bf16_kernel<<<SN, 256, 0, stream>>>(x, wn, hb);

    // 2) weight convert + transpose
    transpose_f32_bf16_kernel<<<dim3(QD / 32, SD / 32), 256, 0, stream>>>(wq, wq_t, SD, QD);
    transpose_f32_bf16_kernel<<<dim3(KD / 32, SD / 32), 256, 0, stream>>>(wk, wk_t, SD, KD);
    transpose_f32_bf16_kernel<<<dim3(KD / 32, SD / 32), 256, 0, stream>>>(wv, wv_t, SD, KD);
    transpose_f32_bf16_kernel<<<dim3(SD / 32, SD / 32), 256, 0, stream>>>(wo, wo_t, SD, SD);

    // 3) projections (WMMA + TDM), RoPE fused into q/k epilogues
    gemm_bf16_wmma_kernel<1><<<dim3(QD / 128, SN / 64), 256, 0, stream>>>(
        hb, wq_t, nullptr, rc, rs, nullptr, q_b, SN, QD, SD);
    gemm_bf16_wmma_kernel<1><<<dim3(KD / 128, SN / 64), 256, 0, stream>>>(
        hb, wk_t, nullptr, rc, rs, nullptr, k_b, SN, KD, SD);
    gemm_bf16_wmma_kernel<2><<<dim3(KD / 128, SN / 64), 256, 0, stream>>>(
        hb, wv_t, nullptr, nullptr, nullptr, nullptr, v_b, SN, KD, SD);

    // 4) WMMA flash attention
    attention_wmma_kernel<<<dim3(NH, SN / 128), 256, 0, stream>>>(q_b, k_b, v_b, a_b);

    // 5) output projection + residual (WMMA + TDM)
    gemm_bf16_wmma_kernel<0><<<dim3(SD / 128, SN / 64), 256, 0, stream>>>(
        a_b, wo_t, x, nullptr, nullptr, out, nullptr, SN, SD, SD);
}
